// DirectionalMamba_33225867002251
// MI455X (gfx1250) — compile-verified, matrix operand order checked by reading, not run on hardware
//
#include <hip/hip_runtime.h>
#include <hip/hip_bf16.h>
#include <math.h>

// ---------------------------------------------------------------------------
// DirectionalMamba on MI455X (gfx1250, wave32, WMMA)
//
// Shapes: Bm=32 (B*D), L=1024 (H*W), C=128, D_INNER=256, DT_RANK=8, D_STATE=32
// Pipeline: LN -> in_proj (WMMA f16) -> depthwise conv+SiLU -> x_proj (WMMA)
//           -> dt softplus -> selective scan (wave32, lane==state)
//           -> gate -> out_proj (WMMA, fused transpose store into d_out)
// ---------------------------------------------------------------------------

typedef __attribute__((ext_vector_type(16))) _Float16 v16h;
typedef __attribute__((ext_vector_type(8)))  _Float16 v8h;
typedef __attribute__((ext_vector_type(8)))  float    v8f;

#define BM     32
#define LSEQ   1024
#define NC     128          // D_MODEL
#define DIN    256          // D_INNER
#define NPROJ  512          // 2*D_INNER
#define DTR    8            // DT_RANK
#define DST    32           // D_STATE
#define MROWS  (BM*LSEQ)    // 32768
#define XDBL_LD 80          // 72 padded to 80 (multiple of 16)

// ---------------- LayerNorm over channels (one row / block) ----------------
__global__ void ln_kernel(const float* __restrict__ x,
                          const float* __restrict__ ln_w,
                          const float* __restrict__ ln_b,
                          _Float16* __restrict__ xn_h) {
  __shared__ float red[NC];
  const int row = blockIdx.x;      // m*1024 + l  (m == depth slice d)
  const int c   = threadIdx.x;     // 0..127
  float v = x[(size_t)c * MROWS + row];
  red[c] = v;
  __syncthreads();
  for (int s = NC / 2; s > 0; s >>= 1) {
    if (c < s) red[c] += red[c + s];
    __syncthreads();
  }
  float mu = red[0] * (1.0f / NC);
  __syncthreads();
  float dv = v - mu;
  red[c] = dv * dv;
  __syncthreads();
  for (int s = NC / 2; s > 0; s >>= 1) {
    if (c < s) red[c] += red[c + s];
    __syncthreads();
  }
  float var = red[0] * (1.0f / NC);
  float xn = dv * rsqrtf(var + 1e-5f) * ln_w[c] + ln_b[c];
  xn_h[(size_t)row * NC + c] = (_Float16)xn;
}

// ---------------- f32 -> f16 weight conversion -----------------------------
__global__ void cvt_f16_kernel(const float* __restrict__ src,
                               _Float16* __restrict__ dst, int n) {
  int i = blockIdx.x * blockDim.x + threadIdx.x;
  if (i < n) dst[i] = (_Float16)src[i];
}

// x_proj_w (72x256) -> zero-padded (80x256) f16
__global__ void cvt_xproj_kernel(const float* __restrict__ src,
                                 _Float16* __restrict__ dst) {
  int i = blockIdx.x * blockDim.x + threadIdx.x;   // 80*256
  if (i >= XDBL_LD * DIN) return;
  int n = i / DIN, k = i % DIN;
  dst[i] = (n < 72) ? (_Float16)src[n * DIN + k] : (_Float16)0.0f;
}

// ---------------- WMMA GEMM: C[M,N] = A[M,K] * W[N,K]^T --------------------
// One wave computes one 16x16 f32 tile with v_wmma_f32_16x16x32_f16.
// A-operand lane layout (16-bit, 16x32): lane&15 = M-row; lane>>4 picks the
// K-chunks {8*hf..8*hf+7} and {16+8*hf..+7}. B mirrors with N instead of M,
// reading contiguous K from the row-major weight (N x K).
// mode 0: row-major store, stride ldc
// mode 1: fused output transpose: row g = m*1024+l -> out[m*128*1024 + n*1024 + l]
__global__ void wmma_gemm_kernel(const _Float16* __restrict__ A,
                                 const _Float16* __restrict__ Bw,
                                 float* __restrict__ Cout,
                                 int M, int N, int K, int ldc, int mode) {
  const int wave = (blockIdx.x * blockDim.x + threadIdx.x) >> 5;
  const int lane = threadIdx.x & 31;
  const int tiles_n = N >> 4;
  const int tm = wave / tiles_n;
  const int tn = wave - tm * tiles_n;
  if (tm * 16 >= M) return;

  const int row = lane & 15;
  const int hf  = lane >> 4;
  const _Float16* pA = A  + (size_t)(tm * 16 + row) * K + 8 * hf;
  const _Float16* pB = Bw + (size_t)(tn * 16 + row) * K + 8 * hf;

  v8f acc = {};
  for (int k0 = 0; k0 < K; k0 += 32) {
    union { v16h v; v8h h[2]; } a, b;
    a.h[0] = *(const v8h*)(pA + k0);
    a.h[1] = *(const v8h*)(pA + k0 + 16);
    b.h[0] = *(const v8h*)(pB + k0);
    b.h[1] = *(const v8h*)(pB + k0 + 16);
    acc = __builtin_amdgcn_wmma_f32_16x16x32_f16(
        /*neg_a=*/false, a.v, /*neg_b=*/false, b.v,
        /*c_mod=*/(short)0, acc, /*reuse_a=*/false, /*reuse_b=*/false);
  }

  const int ncol = tn * 16 + row;
  if (mode == 0) {
#pragma unroll
    for (int r = 0; r < 8; ++r) {
      int Mr = tm * 16 + r + 8 * hf;
      Cout[(size_t)Mr * ldc + ncol] = acc[r];
    }
  } else {
#pragma unroll
    for (int r = 0; r < 8; ++r) {
      int g = tm * 16 + r + 8 * hf;       // = m*1024 + l
      int mrow = g >> 10;
      int l    = g & 1023;
      Cout[(size_t)mrow * (NC * LSEQ) + (size_t)ncol * LSEQ + l] = acc[r];
    }
  }
}

// ---------------- depthwise causal conv(4) + SiLU --------------------------
__global__ void conv_silu_kernel(const float* __restrict__ xz,
                                 const float* __restrict__ cw,
                                 const float* __restrict__ cb,
                                 float* __restrict__ u,
                                 _Float16* __restrict__ u_h) {
  int t = blockIdx.x * blockDim.x + threadIdx.x;   // MROWS*DIN
  if (t >= MROWS * DIN) return;
  int c  = t & (DIN - 1);
  int ml = t >> 8;
  int l  = ml & (LSEQ - 1);
  int mbase = ml - l;                              // m*1024
  float s = cb[c];
#pragma unroll
  for (int j = 0; j < 4; ++j) {
    int lp = l - 3 + j;
    if (lp >= 0) s += cw[c * 4 + j] * xz[(size_t)(mbase + lp) * NPROJ + c];
  }
  float v = s / (1.0f + __expf(-s));               // SiLU
  u[t]   = v;
  u_h[t] = (_Float16)v;
}

// ---------------- dt = softplus(x_dbl[:, :8] @ dt_proj_w^T + b) ------------
__global__ void dt_kernel(const float* __restrict__ x_dbl,
                          const float* __restrict__ dpw,
                          const float* __restrict__ dpb,
                          float* __restrict__ dt) {
  int t = blockIdx.x * blockDim.x + threadIdx.x;   // MROWS*DIN
  if (t >= MROWS * DIN) return;
  int c  = t & (DIN - 1);
  int ml = t >> 8;
  float s = dpb[c];
#pragma unroll
  for (int r = 0; r < DTR; ++r)
    s += x_dbl[(size_t)ml * XDBL_LD + r] * dpw[c * DTR + r];
  dt[t] = (s > 20.0f) ? s : log1pf(__expf(s));
}

// ---------------- selective scan: 1 wave per (m,c), lane == state ----------
__global__ void scan_kernel(const float* __restrict__ dt,
                            const float* __restrict__ u,
                            const float* __restrict__ x_dbl,
                            const float* __restrict__ A_log,
                            const float* __restrict__ Dp,
                            const float* __restrict__ xz,
                            _Float16* __restrict__ y_h) {
  const int wave = (blockIdx.x * blockDim.x + threadIdx.x) >> 5;  // 0..8191
  const int lane = threadIdx.x & 31;                              // state s
  const int m = wave >> 8;
  const int c = wave & (DIN - 1);
  const float Aval = -__expf(A_log[c * DST + lane]);
  const float Dv   = Dp[c];
  float h = 0.0f;
  for (int l = 0; l < LSEQ; ++l) {
    const int base = m * LSEQ + l;
    float dtv = dt[(size_t)base * DIN + c];
    float uv  = u [(size_t)base * DIN + c];
    float bs  = x_dbl[(size_t)base * XDBL_LD + DTR + lane];        // B
    float cs  = x_dbl[(size_t)base * XDBL_LD + DTR + DST + lane];  // C
    h = __expf(dtv * Aval) * h + (dtv * uv) * bs;
    float p = h * cs;
    p += __shfl_xor(p, 16, 32);
    p += __shfl_xor(p, 8, 32);
    p += __shfl_xor(p, 4, 32);
    p += __shfl_xor(p, 2, 32);
    p += __shfl_xor(p, 1, 32);
    if (lane == 0) {
      float z  = xz[(size_t)base * NPROJ + DIN + c];
      float yv = (p + uv * Dv) * (z / (1.0f + __expf(-z)));        // gate
      y_h[(size_t)base * DIN + c] = (_Float16)yv;
    }
  }
}

// ---------------------------------------------------------------------------
extern "C" void kernel_launch(void* const* d_in, const int* in_sizes, int n_in,
                              void* d_out, int out_size, void* d_ws, size_t ws_size,
                              hipStream_t stream) {
  const float* x        = (const float*)d_in[0];
  const float* ln_w     = (const float*)d_in[1];
  const float* ln_b     = (const float*)d_in[2];
  const float* in_pw    = (const float*)d_in[3];   // (512,128)
  const float* conv_w   = (const float*)d_in[4];   // (256,1,4)
  const float* conv_b   = (const float*)d_in[5];
  const float* x_pw     = (const float*)d_in[6];   // (72,256)
  const float* dt_pw    = (const float*)d_in[7];   // (256,8)
  const float* dt_pb    = (const float*)d_in[8];
  const float* A_log    = (const float*)d_in[9];   // (256,32)
  const float* D_param  = (const float*)d_in[10];
  const float* out_pw   = (const float*)d_in[11];  // (128,256)
  float* out = (float*)d_out;

  // ---- workspace layout (256B aligned slabs) ----
  char* p = (char*)d_ws;
  size_t off = 0;
  auto alloc = [&](size_t bytes) {
    void* r = p + off;
    off += (bytes + 255) & ~(size_t)255;
    return r;
  };
  _Float16* xn_h    = (_Float16*)alloc((size_t)MROWS * NC * 2);       // 8 MB
  _Float16* w_in_h  = (_Float16*)alloc((size_t)NPROJ * NC * 2);       // 128 KB
  _Float16* w_xp_h  = (_Float16*)alloc((size_t)XDBL_LD * DIN * 2);    // 40 KB
  _Float16* w_out_h = (_Float16*)alloc((size_t)NC * DIN * 2);         // 64 KB
  float*    xz      = (float*)   alloc((size_t)MROWS * NPROJ * 4);    // 64 MB
  float*    u       = (float*)   alloc((size_t)MROWS * DIN * 4);      // 32 MB
  _Float16* u_h     = (_Float16*)alloc((size_t)MROWS * DIN * 2);      // 16 MB
  float*    x_dbl   = (float*)   alloc((size_t)MROWS * XDBL_LD * 4);  // 10 MB
  float*    dt      = (float*)   alloc((size_t)MROWS * DIN * 4);      // 32 MB
  _Float16* y_h     = (_Float16*)alloc((size_t)MROWS * DIN * 2);      // 16 MB
  (void)ws_size; (void)in_sizes; (void)n_in; (void)out_size;

  // 1) LayerNorm (+transpose to row-major f16)
  ln_kernel<<<MROWS, NC, 0, stream>>>(x, ln_w, ln_b, xn_h);

  // 2) weight converts
  cvt_f16_kernel<<<(NPROJ * NC + 255) / 256, 256, 0, stream>>>(in_pw, w_in_h, NPROJ * NC);
  cvt_xproj_kernel<<<(XDBL_LD * DIN + 255) / 256, 256, 0, stream>>>(x_pw, w_xp_h);
  cvt_f16_kernel<<<(NC * DIN + 255) / 256, 256, 0, stream>>>(out_pw, w_out_h, NC * DIN);

  // 3) in_proj: (32768 x 512) = xn (32768 x 128) @ W^T ; tiles 2048 x 32
  {
    int waves = (MROWS / 16) * (NPROJ / 16);
    wmma_gemm_kernel<<<waves * 32 / 256, 256, 0, stream>>>(
        xn_h, w_in_h, xz, MROWS, NPROJ, NC, NPROJ, 0);
  }

  // 4) depthwise causal conv + SiLU -> u
  conv_silu_kernel<<<(MROWS * DIN) / 256, 256, 0, stream>>>(xz, conv_w, conv_b, u, u_h);

  // 5) x_proj: (32768 x 80pad) = u (32768 x 256) @ W^T ; tiles 2048 x 5
  {
    int waves = (MROWS / 16) * (XDBL_LD / 16);
    wmma_gemm_kernel<<<waves * 32 / 256, 256, 0, stream>>>(
        u_h, w_xp_h, x_dbl, MROWS, XDBL_LD, DIN, XDBL_LD, 0);
  }

  // 6) dt softplus (K=8, VALU)
  dt_kernel<<<(MROWS * DIN) / 256, 256, 0, stream>>>(x_dbl, dt_pw, dt_pb, dt);

  // 7) selective scan + skip + gate -> y_h (8192 waves, lane = state)
  scan_kernel<<<(BM * DIN * 32) / 256, 256, 0, stream>>>(
      dt, u, x_dbl, A_log, D_param, xz, y_h);

  // 8) out_proj with fused channel-major transpose store into d_out
  {
    int waves = (MROWS / 16) * (NC / 16);
    wmma_gemm_kernel<<<waves * 32 / 256, 256, 0, stream>>>(
        y_h, w_out_h, out, MROWS, NC, DIN, 0, 1);
  }
}
